// MyModel_54382875902169
// MI455X (gfx1250) — matrix operand order repaired and checked by reading, not compiled, not run on hardware
//
#include <hip/hip_runtime.h>
#include <hip/hip_bf16.h>

typedef __attribute__((ext_vector_type(16))) _Float16 v16h;
typedef __attribute__((ext_vector_type(4)))  _Float16 v4h;
typedef __attribute__((ext_vector_type(2)))  _Float16 v2h;
typedef __attribute__((ext_vector_type(8)))  float    v8f;

#define DM   256    // model dim
#define NH   8      // heads
#define HD   32     // head dim
#define FF   1024
#define BB   32     // batch
#define SS   256    // seq len
#define TT   (BB*SS)  // tokens = 8192

// ---------------------------------------------------------------------------
// Fragment-ready LDS layouts (inverse of gfx1250 wave32 WMMA layouts).
// A-tile (16x32, MxK): (m,k) -> lane = m + 16*((k>>3)&1), h = (k&7) + 8*(k>>4)
// B-tile (32x16, KxN): (k,n) -> lane = n + 16*((k>>3)&1), h = (k&7) + 8*(k>>4)
// For a 4-aligned quad of consecutive k: lane is constant, h is 4 consecutive
// halves -> one packed ds_store_b64 / ds_load_b64.
// ---------------------------------------------------------------------------
__device__ __forceinline__ int a_lane(int m, int k) { return m + ((k & 8) ? 16 : 0); }
__device__ __forceinline__ int a_half(int k)        { return (k & 7) + ((k & 16) ? 8 : 0); }
__device__ __forceinline__ int b_lane(int n, int k) { return n + ((k & 8) ? 16 : 0); }
__device__ __forceinline__ int b_half(int k)        { return (k & 7) + ((k & 16) ? 8 : 0); }

__device__ __forceinline__ v4h pack4(float a0, float a1, float a2, float a3) {
  v4h q;
  q.lo = (v2h)__builtin_amdgcn_cvt_pkrtz(a0, a1);
  q.hi = (v2h)__builtin_amdgcn_cvt_pkrtz(a2, a3);
  return q;
}

// ---------------------------------------------------------------------------
// GEMM: C[M,N] = act(A[M,K] * W[K,N] + bias), row-major fp32 in/out,
// fp16 WMMA math. BM=BN=BK=64, 256 threads (8 waves, 2 C-tiles each).
// Quad-granular fragment staging: b128 global loads, pk-cvt, b64 LDS stores.
// ---------------------------------------------------------------------------
#define BM 64
#define BN 64
#define BK 64

__global__ __launch_bounds__(256) void gemm_bias_kernel(
    const float* __restrict__ A, const float* __restrict__ W,
    const float* __restrict__ bias, float* __restrict__ C,
    int M, int N, int K, int relu)
{
  __shared__ alignas(32) _Float16 As_f[4][2][32][16];  // [tm][kc][lane][h] 8KB
  __shared__ alignas(32) _Float16 Bs_f[4][2][32][16];  // [tn][kc][lane][h] 8KB

  const int tid  = threadIdx.x;
  const int wave = tid >> 5;
  const int lane = tid & 31;

  const int bm = blockIdx.y * BM;
  const int bn = blockIdx.x * BN;

  const int tm  = wave >> 1;        // 0..3
  const int tn0 = (wave & 1) * 2;   // 0 or 2

  v8f acc0 = {};
  v8f acc1 = {};

  for (int k0 = 0; k0 < K; k0 += BK) {
    // stage A block: 64 rows x 16 k-quads = 1024 quads, 4 per thread.
#pragma unroll
    for (int rr = 0; rr < 4; ++rr) {
      int qi = rr * 256 + tid;
      int r  = qi >> 4;             // row 0..63
      int k  = (qi & 15) * 4;       // 4-aligned k
      float4 f = *(const float4*)&A[(size_t)(bm + r) * K + k0 + k];
      *(v4h*)&As_f[r >> 4][k >> 5][a_lane(r & 15, k)][a_half(k)] =
          pack4(f.x, f.y, f.z, f.w);
    }
    // stage W block: 16 k-quads x 64 n = 1024 quads, 4 per thread.
    // n is the fast (coalescing) axis; 4 strided-row loads per quad.
#pragma unroll
    for (int rr = 0; rr < 4; ++rr) {
      int qi = rr * 256 + tid;
      int n  = qi & 63;
      int k  = (qi >> 6) * 4;       // 4-aligned k
      const float* wp = &W[(size_t)(k0 + k) * N + bn + n];
      *(v4h*)&Bs_f[n >> 4][k >> 5][b_lane(n & 15, k)][b_half(k)] =
          pack4(wp[0], wp[N], wp[2 * (size_t)N], wp[3 * (size_t)N]);
    }
    __syncthreads();

#pragma unroll
    for (int kc = 0; kc < 2; ++kc) {
      v16h af  = *(const v16h*)&As_f[tm][kc][lane][0];
      v16h bf0 = *(const v16h*)&Bs_f[tn0][kc][lane][0];
      v16h bf1 = *(const v16h*)&Bs_f[tn0 + 1][kc][lane][0];
      acc0 = __builtin_amdgcn_wmma_f32_16x16x32_f16(false, af, false, bf0,
                                                    (short)0, acc0, false, false);
      acc1 = __builtin_amdgcn_wmma_f32_16x16x32_f16(false, af, false, bf1,
                                                    (short)0, acc1, false, false);
    }
    __syncthreads();
  }

  const int rbase = bm + tm * 16 + ((lane & 16) ? 8 : 0);
  const int c0    = bn + tn0 * 16 + (lane & 15);
  const int c1    = c0 + 16;
  const float b0  = bias[c0];
  const float b1  = bias[c1];
#pragma unroll
  for (int r = 0; r < 8; ++r) {
    int row  = rbase + r;
    float v0 = acc0[r] + b0;
    float v1 = acc1[r] + b1;
    if (relu) { v0 = fmaxf(v0, 0.0f); v1 = fmaxf(v1, 0.0f); }
    C[(size_t)row * N + c0] = v0;
    C[(size_t)row * N + c1] = v1;
  }
}

// ---------------------------------------------------------------------------
// Fused attention: one block per (batch, head, 32-query block).
// scores = (q k^T)/sqrt(HD) * exp(-gamma*|dq-dk|), causal mask, softmax,
// out = probs @ v.  All matmuls on WMMA; all fragment traffic vectorized.
// maskk==0: allow k<=q (tril); maskk==1: allow k<q (tril,-1).
// ---------------------------------------------------------------------------
__global__ __launch_bounds__(256) void attn_kernel(
    const float* __restrict__ Q, const float* __restrict__ K,
    const float* __restrict__ V, const float* __restrict__ gammas,
    float* __restrict__ O, int maskk)
{
  __shared__ alignas(32) _Float16 Qf[2][32][16];      // q A-tiles        (2 KB)
  __shared__ alignas(32) _Float16 KVf[16][32][16];    // K^T then V tiles (16 KB)
  __shared__ alignas(32) _Float16 Pf[2][8][32][16];   // scores/probs, A-layout (16 KB)

  const int tid  = threadIdx.x;
  const int wave = tid >> 5;
  const int lane = tid & 31;

  const int qb = blockIdx.x & 7;          // query block (8 per head)
  const int bh = blockIdx.x >> 3;
  const int h  = bh & (NH - 1);
  const int b  = bh / NH;

  const float gamma = gammas[h];
  const float scale = 0.17677669529663687f;   // 1/sqrt(32)

  // ---- stage Q (A-layout) and K^T (B-layout), quad-granular ---------------
  {   // Q: 32 rows x 8 d-quads = 256 quads, 1 per thread
    int row = tid >> 3;
    int d   = (tid & 7) * 4;
    float4 f = *(const float4*)&Q[(size_t)(b * SS + qb * 32 + row) * DM + h * HD + d];
    *(v4h*)&Qf[row >> 4][a_lane(row & 15, d)][a_half(d)] = pack4(f.x, f.y, f.z, f.w);
  }
#pragma unroll
  for (int rr = 0; rr < 8; ++rr) {   // K: 256 keys x 8 d-quads = 2048 quads
    int qi  = rr * 256 + tid;
    int key = qi >> 3;
    int d   = (qi & 7) * 4;
    float4 f = *(const float4*)&K[(size_t)(b * SS + key) * DM + h * HD + d];
    *(v4h*)&KVf[key >> 4][b_lane(key & 15, d)][b_half(d)] = pack4(f.x, f.y, f.z, f.w);
  }
  __syncthreads();

  // ---- phase 1: scores + fused scale/decay/mask, stored f16 in A-layout ---
  {
    const int rm  = wave >> 2;            // 0..1 (query row tile)
    const int ct0 = (wave & 3) * 4;       // 4 key tiles per wave
    v16h af = *(const v16h*)&Qf[rm][lane][0];
#pragma unroll
    for (int t = 0; t < 4; ++t) {
      const int ct = ct0 + t;
      v16h bf = *(const v16h*)&KVf[ct][lane][0];
      v8f acc = {};
      acc = __builtin_amdgcn_wmma_f32_16x16x32_f16(false, af, false, bf,
                                                   (short)0, acc, false, false);
      const int mhi = (lane & 16) ? 8 : 0;
      const int cc  = ct * 16 + (lane & 15);          // key position
      const int k   = cc & 31;
      _Float16* pb  = &Pf[rm][cc >> 5][0][a_half(k)];
      const int lhi = (k & 8) ? 16 : 0;
#pragma unroll
      for (int r = 0; r < 8; ++r) {
        const int rr    = rm * 16 + r + mhi;          // local q row (0..31)
        const int pos_q = qb * 32 + rr;
        float s = acc[r] * scale;
        s *= __expf(-gamma * fabsf((float)(pos_q - cc)));
        bool ok = maskk ? (cc < pos_q) : (cc <= pos_q);
        s = ok ? s : -10000.0f;
        pb[((rr & 15) + lhi) * 16] = (_Float16)s;
      }
    }
  }
  __syncthreads();

  // ---- phase 2: stage V (B-layout) + parallel row softmax -----------------
#pragma unroll
  for (int rr = 0; rr < 8; ++rr) {   // V: 64 key-quads x 32 d = 2048 quads
    int qi  = rr * 256 + tid;
    int d   = qi & 31;               // coalescing axis across threads
    int key = (qi >> 5) * 4;         // 4-aligned key
    const float* vp = &V[(size_t)(b * SS + key) * DM + h * HD + d];
    // tile = (key/32)*2 + (d/16); within tile: K-dim = key&31, N = d&15
    *(v4h*)&KVf[(key >> 5) * 2 + (d >> 4)]
              [b_lane(d & 15, key & 31)][b_half(key & 31)] =
        pack4(vp[0], vp[DM], vp[2 * DM], vp[3 * DM]);
  }
  {   // softmax: 8 threads per row; each owns one kt tile (32 cols, 8 quads)
    const int r   = tid >> 3;        // row 0..31
    const int kt  = tid & 7;         // this thread's 32-column chunk
    const int m   = r & 15;
    const int om  = r >> 4;
    _Float16* base = &Pf[om][kt][0][0];

    float mx = -3.4e38f;
#pragma unroll
    for (int jq = 0; jq < 8; ++jq) {
      int k = jq * 4;
      v4h q = *(const v4h*)&base[a_lane(m, k) * 16 + a_half(k)];
      mx = fmaxf(mx, fmaxf(fmaxf((float)q.x, (float)q.y),
                           fmaxf((float)q.z, (float)q.w)));
    }
#pragma unroll
    for (int off = 1; off < 8; off <<= 1) mx = fmaxf(mx, __shfl_xor(mx, off, 32));

    float sum = 0.0f;
#pragma unroll
    for (int jq = 0; jq < 8; ++jq) {
      int k = jq * 4;
      v4h* p = (v4h*)&base[a_lane(m, k) * 16 + a_half(k)];
      v4h q = *p;
      float e0 = __expf((float)q.x - mx);
      float e1 = __expf((float)q.y - mx);
      float e2 = __expf((float)q.z - mx);
      float e3 = __expf((float)q.w - mx);
      sum += (e0 + e1) + (e2 + e3);
      *p = pack4(e0, e1, e2, e3);
    }
#pragma unroll
    for (int off = 1; off < 8; off <<= 1) sum += __shfl_xor(sum, off, 32);

    const float inv = 1.0f / sum;
#pragma unroll
    for (int jq = 0; jq < 8; ++jq) {
      int k = jq * 4;
      v4h* p = (v4h*)&base[a_lane(m, k) * 16 + a_half(k)];
      v4h q = *p;
      *p = pack4((float)q.x * inv, (float)q.y * inv,
                 (float)q.z * inv, (float)q.w * inv);
    }
  }
  __syncthreads();

  // ---- phase 3: out = probs @ V (4 tiles of 16x16, K over 256 keys) -------
  if (wave < 4) {
    const int om = wave >> 1;   // query row tile
    const int on = wave & 1;    // head-dim col tile
    v8f acc = {};
#pragma unroll
    for (int kt = 0; kt < 8; ++kt) {
      v16h af = *(const v16h*)&Pf[om][kt][lane][0];
      v16h bf = *(const v16h*)&KVf[kt * 2 + on][lane][0];
      acc = __builtin_amdgcn_wmma_f32_16x16x32_f16(false, af, false, bf,
                                                   (short)0, acc, false, false);
    }
    const int rbase = b * SS + qb * 32 + om * 16 + ((lane & 16) ? 8 : 0);
    const int col   = h * HD + on * 16 + (lane & 15);
#pragma unroll
    for (int r = 0; r < 8; ++r)
      O[(size_t)(rbase + r) * DM + col] = acc[r];
  }
}

// ---------------------------------------------------------------------------
// out = LayerNorm(X + Yp) * g + b  -- one block (256 thr = 8 waves) per row,
// wave32 shuffle reductions + one cross-wave combine.
// ---------------------------------------------------------------------------
__global__ __launch_bounds__(256) void resln_kernel(
    const float* __restrict__ X, const float* __restrict__ Yp,
    const float* __restrict__ g, const float* __restrict__ bta,
    float* __restrict__ Out)
{
  __shared__ float partial[8];
  __shared__ float stats[2];
  const int row  = blockIdx.x;
  const int d    = threadIdx.x;
  const int wave = d >> 5;
  const int lane = d & 31;

  float v = X[(size_t)row * DM + d] + Yp[(size_t)row * DM + d];

  float s = v;
#pragma unroll
  for (int off = 16; off > 0; off >>= 1) s += __shfl_xor(s, off, 32);
  if (lane == 0) partial[wave] = s;
  __syncthreads();
  if (d == 0) {
    float t = 0.0f;
#pragma unroll
    for (int i = 0; i < 8; ++i) t += partial[i];
    stats[0] = t * (1.0f / DM);
  }
  __syncthreads();
  const float mean = stats[0];
  const float dv   = v - mean;

  float s2 = dv * dv;
#pragma unroll
  for (int off = 16; off > 0; off >>= 1) s2 += __shfl_xor(s2, off, 32);
  if (lane == 0) partial[wave] = s2;
  __syncthreads();
  if (d == 0) {
    float t = 0.0f;
#pragma unroll
    for (int i = 0; i < 8; ++i) t += partial[i];
    stats[1] = t * (1.0f / DM);
  }
  __syncthreads();

  Out[(size_t)row * DM + d] = dv * rsqrtf(stats[1] + 1e-5f) * g[d] + bta[d];
}

// ---------------------------------------------------------------------------
// Embedding gather; combined=1 uses idx = x + x*resp (decoder inputs)
// ---------------------------------------------------------------------------
__global__ __launch_bounds__(256) void embed_kernel(
    const int* __restrict__ xf, const int* __restrict__ resp,
    const float* __restrict__ table, float* __restrict__ out,
    int feat, int combined)
{
  const int gid = blockIdx.x * 256 + threadIdx.x;   // T*D threads
  const int t = gid >> 8;
  const int d = gid & 255;
  const int bb = t >> 8;          // t / S
  const int s  = t & 255;         // t % S
  int x = xf[bb * 3 * SS + feat * SS + s];
  int idx = combined ? (x + x * resp[bb * SS + s]) : x;
  out[(size_t)t * DM + d] = table[(size_t)idx * DM + d];
}

// flow accumulate: dst = init ? src : dst + src
__global__ __launch_bounds__(256) void accum_kernel(
    float* __restrict__ dst, const float* __restrict__ src, int init, int n)
{
  int i = blockIdx.x * 256 + threadIdx.x;
  if (i < n) dst[i] = init ? src[i] : (dst[i] + src[i]);
}

// predict head: out[t] = dot(X[t,:], wp) + bp   (one wave per token)
__global__ __launch_bounds__(256) void dotp_kernel(
    const float* __restrict__ X, const float* __restrict__ wp,
    const float* __restrict__ bp, float* __restrict__ out)
{
  const int wave = threadIdx.x >> 5;
  const int lane = threadIdx.x & 31;
  const int t = blockIdx.x * 8 + wave;
  const float* row = X + (size_t)t * DM;
  float s = 0.0f;
#pragma unroll
  for (int i = 0; i < 8; ++i) s += row[lane + i * 32] * wp[lane + i * 32];
#pragma unroll
  for (int off = 16; off > 0; off >>= 1) s += __shfl_xor(s, off, 32);
  if (lane == 0) out[t] = s + bp[0];
}

// ---------------------------------------------------------------------------
// Host orchestration
// ---------------------------------------------------------------------------
extern "C" void kernel_launch(void* const* d_in, const int* in_sizes, int n_in,
                              void* d_out, int out_size, void* d_ws, size_t ws_size,
                              hipStream_t stream) {
  (void)in_sizes; (void)n_in; (void)out_size; (void)ws_size;

  auto F = [&](int i) { return (const float*)d_in[i]; };
  const int* x_feature  = (const int*)d_in[0];
  const int* x_response = (const int*)d_in[1];

  // param indices (setup_inputs dict order, nested dicts flattened in order)
  const int encB[3]  = {8, 19, 30};     // p_enc, s_enc, q_enc  (11 each)
  const int ffB[3]   = {41, 45, 49};    // p_ff, s_ff, q_ff     (4 each)
  const int dec1B[3] = {53, 64, 75};
  const int dec2B[3] = {86, 97, 108};
  const int predB    = 119;             // w1,b1,w2,b2,wp,bp

  const size_t TD = (size_t)TT * DM;    // 8192*256
  const size_t TH = (size_t)TT * FF;    // 8192*1024

  float* ws      = (float*)d_ws;
  float* bufA    = ws;                 // embeddings / generic input
  float* bufQ    = bufA + TD;
  float* bufK    = bufQ + TD;
  float* bufV    = bufK + TD;
  float* bufO    = bufV + TD;          // attention out (pre-wo)
  float* bufP    = bufO + TD;          // wo projection / ffn2 out
  float* bufX    = bufP + TD;          // encoder-stream keep
  float* bufY    = bufX + TD;          // decoder1 keep
  float* bufT    = bufY + TD;          // mha output temp
  float* bufH    = bufT + TD;          // ffn hidden (T x FF)
  float* bufFlow = bufH + TH;

  auto gemm = [&](const float* A, const float* W, const float* bias, float* C,
                  int M, int N, int Kd, int relu) {
    dim3 grid(N / BN, M / BM);
    gemm_bias_kernel<<<grid, 256, 0, stream>>>(A, W, bias, C, M, N, Kd, relu);
  };

  auto run_mha = [&](int base, const float* Qsrc, const float* Ksrc,
                     const float* Vsrc, int maskk, float* out) {
    gemm(Qsrc, F(base + 0), F(base + 1), bufQ, TT, DM, DM, 0);
    gemm(Ksrc, F(base + 2), F(base + 3), bufK, TT, DM, DM, 0);
    gemm(Vsrc, F(base + 4), F(base + 5), bufV, TT, DM, DM, 0);
    attn_kernel<<<BB * NH * 8, 256, 0, stream>>>(bufQ, bufK, bufV,
                                                 F(base + 8), bufO, maskk);
    gemm(bufO, F(base + 6), F(base + 7), bufP, TT, DM, DM, 0);
    resln_kernel<<<TT, 256, 0, stream>>>(Qsrc, bufP, F(base + 9), F(base + 10), out);
  };

  auto run_ffn = [&](int base, const float* in, float* out) {
    gemm(in,   F(base + 0), F(base + 1), bufH, TT, FF, DM, 1);
    gemm(bufH, F(base + 2), F(base + 3), out,  TT, DM, FF, 1);
  };

  const int embGrid = (int)(TD / 256);

  for (int i = 0; i < 3; ++i) {
    // encoder: e_i -> mha(tril) -> ffn
    embed_kernel<<<embGrid, 256, 0, stream>>>(x_feature, x_response,
                                              F(2 + i), bufA, i, 0);
    run_mha(encB[i], bufA, bufA, bufA, /*maskk=*/0, bufT);
    run_ffn(ffB[i], bufT, bufX);

    // decoder1: combined embedding -> mha(tril)
    embed_kernel<<<embGrid, 256, 0, stream>>>(x_feature, x_response,
                                              F(5 + i), bufA, i, 1);
    run_mha(dec1B[i], bufA, bufA, bufA, /*maskk=*/0, bufY);

    // decoder2: Q=K=x, V=y, strict causal mask (tril, k=-1)
    run_mha(dec2B[i], bufX, bufX, bufY, /*maskk=*/1, bufT);

    accum_kernel<<<(int)((TD + 255) / 256), 256, 0, stream>>>(
        bufFlow, bufT, i == 0 ? 1 : 0, (int)TD);
  }

  // predict head: relu(relu(flow@w1+b1)@w2+b2) @ wp + bp
  gemm(bufFlow, F(predB + 0), F(predB + 1), bufH, TT, FF, DM, 1);
  gemm(bufH,    F(predB + 2), F(predB + 3), bufP, TT, DM, FF, 1);
  dotp_kernel<<<TT / 8, 256, 0, stream>>>(bufP, F(predB + 4), F(predB + 5),
                                          (float*)d_out);
}
